// Attention_48052094107920
// MI455X (gfx1250) — matrix operand order repaired and checked by reading, not compiled, not run on hardware
//
#include <hip/hip_runtime.h>

// ---------------------------------------------------------------------------
// CDNA5 / gfx1250 fused attention pipeline (wave32, WMMA f32_16x16x32_f16,
// async global->LDS staging, global prefetch)
//   B=8, N=577, C=768, GH=3, LH=12, HD=64, ML=3, Np=608 (=19*32)
// Kernels:
//   1) k_cvt_x      : x f32 -> f16
//   2) k_build_wT   : [Wq|Wk|Wv] -> f16 transposed [1152][768]
//   3) k_build_pT   : proj_w -> f16 transposed [768][768]
//   4) k_qkv        : WMMA GEMM -> Q[b][g][q][d], K[b][g][q][d], Vt[b][h][d][q]
//   5) k_attn       : fused score/mask-mix/online-softmax/PV (12 waves=12 heads)
//                     masks staged via GLOBAL_LOAD_ASYNC_TO_LDS_B32 overlapped
//                     with the QK^T WMMAs, fenced by s_wait_asynccnt.
//   6) k_proj       : WMMA GEMM + bias -> f32 out
// ---------------------------------------------------------------------------

typedef _Float16 half_t;
typedef __attribute__((ext_vector_type(16))) _Float16 v16h;
typedef __attribute__((ext_vector_type(8)))  _Float16 v8h;
typedef __attribute__((ext_vector_type(8)))  float    v8f;

#define B_    8
#define NTOK  577
#define C_    768
#define LH_   12
#define GH_   3
#define ML_   3
#define HD_   64
#define NP_   608
#define QT_   37          // attention q-tiles (covers 0..591)
#define QTF_  38          // qkv row tiles (fills all 608 padded rows)
#define KCH_  19          // 608 / 32 key chunks
#define QKVO_ 1152
#define NEG_BIG (-1.0e30f)

static __device__ __forceinline__ v8f wmma16(v16h a, v16h b, v8f c) {
  // D = A(16x32 f16) * B(32x16 f16) + C(16x16 f32)
  return __builtin_amdgcn_wmma_f32_16x16x32_f16(false, a, false, b, (short)0, c,
                                                false, false);
}

// A-fragment: row m, K pattern per ISA 7.12.2 (16-bit A 16x32):
//   lanes<16 : K = {kk+0..7, kk+16..23}; lanes>=16: +8
static __device__ __forceinline__ v16h load_a16(const half_t* __restrict__ base,
                                                int stride, int m, int kk, int hi) {
  const half_t* p = base + (size_t)m * stride + kk + hi * 8;
  v8h lo = *(const v8h*)(p);
  v8h hh = *(const v8h*)(p + 16);
  v16h a;
#pragma unroll
  for (int i = 0; i < 8; ++i) { a[i] = lo[i]; a[i + 8] = hh[i]; }
  return a;
}

// B-fragment: src stored as SRC[col][k] (k contiguous).
// lanes<16 hold K={kk+0..7, kk+16..23} of column col, lanes>=16: +8 (matches A).
static __device__ __forceinline__ v16h load_b16(const half_t* __restrict__ base,
                                                int stride, int col, int kk, int hi) {
  const half_t* p = base + (size_t)col * stride + kk + hi * 8;
  v8h lo = *(const v8h*)(p);
  v8h hh = *(const v8h*)(p + 16);
  v16h b;
#pragma unroll
  for (int i = 0; i < 8; ++i) { b[i] = lo[i]; b[i + 8] = hh[i]; }
  return b;
}

// async gather of one f32 from global into LDS (ASYNCcnt-tracked, no VGPR data)
static __device__ __forceinline__ void async_ld_lds_b32(void* lds_dst,
                                                        const void* gsrc) {
  unsigned ldsoff = (unsigned)(uintptr_t)lds_dst;      // addr[31:0] = LDS offset
  unsigned long long ga = (unsigned long long)(uintptr_t)gsrc;
  asm volatile("global_load_async_to_lds_b32 %0, %1, off"
               :: "v"(ldsoff), "v"(ga)
               : "memory");
}

static __device__ __forceinline__ void wait_asynccnt0() {
  asm volatile("s_wait_asynccnt 0x0" ::: "memory");
}

// ------------------------------ conversions -------------------------------

__global__ void k_cvt_x(const float* __restrict__ x, half_t* __restrict__ xh, int n) {
  int i = blockIdx.x * blockDim.x + threadIdx.x;
  if (i < n) xh[i] = (half_t)x[i];
}

__global__ void k_build_wT(const float* __restrict__ Wq, const float* __restrict__ Wk,
                           const float* __restrict__ Wv, half_t* __restrict__ WT) {
  int i = blockIdx.x * blockDim.x + threadIdx.x;      // i < 1152*768
  if (i >= QKVO_ * C_) return;
  int oc = i / C_, kin = i % C_;
  float v;
  if (oc < 192)      v = Wq[(size_t)kin * 192 + oc];
  else if (oc < 384) v = Wk[(size_t)kin * 192 + (oc - 192)];
  else               v = Wv[(size_t)kin * C_ + (oc - 384)];
  WT[(size_t)oc * C_ + kin] = (half_t)v;
}

__global__ void k_build_pT(const float* __restrict__ Pw, half_t* __restrict__ PT) {
  int i = blockIdx.x * blockDim.x + threadIdx.x;      // i < 768*768
  if (i >= C_ * C_) return;
  int oc = i / C_, kin = i % C_;
  PT[(size_t)oc * C_ + kin] = (half_t)Pw[(size_t)kin * C_ + oc];
}

// ------------------------------ QKV GEMM ----------------------------------
// grid (QTF_, B_), 256 threads (8 waves). Wave w owns 9 col-tiles of 16.
__global__ __launch_bounds__(256) void k_qkv(const half_t* __restrict__ xh,
                                             const half_t* __restrict__ WT,
                                             half_t* __restrict__ Qw,
                                             half_t* __restrict__ Kw,
                                             half_t* __restrict__ Vt) {
  const int qt = blockIdx.x, b = blockIdx.y;
  const int tid = threadIdx.x, wave = tid >> 5, lane = tid & 31;
  const int l16 = lane & 15, hi = lane >> 4;

  int q_a = qt * 16 + l16;                 // A-row (this lane) in padded space
  int gr  = b * NTOK + (q_a > 576 ? 576 : q_a);   // clamp pad rows

  v8f acc[9];
#pragma unroll
  for (int j = 0; j < 9; ++j) acc[j] = (v8f){0,0,0,0,0,0,0,0};

  for (int kk = 0; kk < C_; kk += 32) {
    v16h a = load_a16(xh, C_, gr, kk, hi);
#pragma unroll
    for (int j = 0; j < 9; ++j) {
      int ct = wave * 9 + j;
      v16h bb = load_b16(WT, C_, ct * 16 + l16, kk, hi);
      acc[j] = wmma16(a, bb, acc[j]);
    }
  }
  // scatter (C layout: col = lane%16, row r -> M = r + 8*hi)
#pragma unroll
  for (int j = 0; j < 9; ++j) {
    int oc = (wave * 9 + j) * 16 + l16;
#pragma unroll
    for (int r = 0; r < 8; ++r) {
      int q = qt * 16 + r + 8 * hi;                       // < 608
      half_t v = (half_t)acc[j][r];
      if (oc < 192) {
        Qw[(((size_t)b * GH_ + (oc >> 6)) * NP_ + q) * HD_ + (oc & 63)] = v;
      } else if (oc < 384) {
        int o = oc - 192;
        Kw[(((size_t)b * GH_ + (o >> 6)) * NP_ + q) * HD_ + (o & 63)] = v;
      } else {
        int o = oc - 384;                                  // V stored transposed
        Vt[(((size_t)b * LH_ + (o >> 6)) * HD_ + (o & 63)) * NP_ + q] = v;
      }
    }
  }
}

// --------------------------- fused attention ------------------------------
// grid (QT_, B_), 384 threads = 12 waves; wave h handles local head h.
__global__ __launch_bounds__(384) void k_attn(const half_t* __restrict__ Qw,
                                              const half_t* __restrict__ Kw,
                                              const half_t* __restrict__ Vt,
                                              const float* __restrict__ masks,
                                              const float* __restrict__ mask_proj,
                                              half_t* __restrict__ Of16) {
  const int qt = blockIdx.x, b = blockIdx.y;
  const int tid = threadIdx.x, wave = tid >> 5, lane = tid & 31;
  const int l16 = lane & 15, hi = lane >> 4;
  const int h = wave;

  __shared__ __align__(16) float  S_lds[GH_][16 * 32];   // [g][col*16 + row]
  __shared__ __align__(16) float  M_lds[ML_][16 * 32];   // [l][row*32 + col]
  __shared__ __align__(16) half_t P_lds[LH_][16 * 32];   // per-wave P tile

  // per-head mix coefficients, SCALE folded in
  float cml[ML_][GH_];
#pragma unroll
  for (int l = 0; l < ML_; ++l)
#pragma unroll
    for (int g = 0; g < GH_; ++g)
      cml[l][g] = 0.125f * mask_proj[l * (GH_ * LH_) + g * LH_ + h];

  float mrun[8], lrun[8];
  v8f Oacc[4];
#pragma unroll
  for (int r = 0; r < 8; ++r) { mrun[r] = -3.0e38f; lrun[r] = 0.0f; }
#pragma unroll
  for (int n = 0; n < 4; ++n) Oacc[n] = (v8f){0,0,0,0,0,0,0,0};

  const half_t* Vh = Vt + ((size_t)b * LH_ + h) * HD_ * NP_;

  for (int ch = 0; ch < KCH_; ++ch) {
    __syncthreads();   // protect S/M LDS reuse from previous chunk

    // ---- async-stage mask patch [ML][16 rows][32 cols] into LDS ----------
    // issued now, consumed after the barrier; overlaps with the QK^T WMMAs
    for (int i = tid; i < ML_ * 16 * 32; i += 384) {
      int l = i >> 9, rr = (i >> 5) & 15, cc = i & 31;
      int q = qt * 16 + rr, k = ch * 32 + cc;
      float* dst = &M_lds[l][rr * 32 + cc];
      if (q < NTOK && k < NTOK) {
        async_ld_lds_b32(dst, masks + ((size_t)q * NTOK + k) * ML_ + l);
      } else {
        *dst = 0.0f;
      }
    }

    // prefetch next chunk's V columns for this head (global_prefetch_b8)
    if (ch + 1 < KCH_) {
      __builtin_prefetch(Vh + (size_t)(l16 * 4) * NP_ + (ch + 1) * 32, 0, 1);
    }

    // waves 0..2: S_g = Q_g . K_g^T for this 16x32 chunk
    if (wave < GH_) {
      int g = wave;
      const half_t* Qg = Qw + (((size_t)b * GH_ + g) * NP_ + qt * 16) * HD_;
      const half_t* Kg = Kw + ((size_t)b * GH_ + g) * NP_ * HD_;
#pragma unroll
      for (int ct = 0; ct < 2; ++ct) {
        v8f s = (v8f){0,0,0,0,0,0,0,0};
#pragma unroll
        for (int ks = 0; ks < 2; ++ks) {
          v16h a  = load_a16(Qg, HD_, l16, ks * 32, hi);
          v16h bb = load_b16(Kg, HD_, ch * 32 + ct * 16 + l16, ks * 32, hi);
          s = wmma16(a, bb, s);
        }
#pragma unroll
        for (int r = 0; r < 8; ++r)
          S_lds[g][(ct * 16 + l16) * 16 + r + 8 * hi] = s[r];
      }
    }
    wait_asynccnt0();  // this wave's async LDS writes are now visible
    __syncthreads();

    // mix + online softmax (rows r+8*hi live in this half-wave, cols across 16 lanes)
    float p[2][8], amax[8];
#pragma unroll
    for (int r = 0; r < 8; ++r) amax[r] = -3.0e38f;
#pragma unroll
    for (int t = 0; t < 2; ++t) {
      int col = t * 16 + l16;
      int k   = ch * 32 + col;
      bool kok = (k < NTOK);
#pragma unroll
      for (int r = 0; r < 8; ++r) {
        int row = r + 8 * hi;
        float s0 = S_lds[0][col * 16 + row];
        float s1 = S_lds[1][col * 16 + row];
        float s2 = S_lds[2][col * 16 + row];
        float a = 0.0f;
#pragma unroll
        for (int l = 0; l < ML_; ++l)
          a += M_lds[l][row * 32 + col] *
               (s0 * cml[l][0] + s1 * cml[l][1] + s2 * cml[l][2]);
        a = kok ? a : NEG_BIG;
        p[t][r] = a;
        amax[r] = fmaxf(amax[r], a);
      }
    }
#pragma unroll
    for (int r = 0; r < 8; ++r) {
#pragma unroll
      for (int off = 1; off < 16; off <<= 1)
        amax[r] = fmaxf(amax[r], __shfl_xor(amax[r], off, 32));
    }
    float alpha[8], rsum[8];
#pragma unroll
    for (int r = 0; r < 8; ++r) {
      float mn = fmaxf(mrun[r], amax[r]);
      alpha[r] = __expf(mrun[r] - mn);
      mrun[r]  = mn;
      rsum[r]  = 0.0f;
    }
#pragma unroll
    for (int t = 0; t < 2; ++t)
#pragma unroll
      for (int r = 0; r < 8; ++r) {
        p[t][r] = __expf(p[t][r] - mrun[r]);
        rsum[r] += p[t][r];
      }
#pragma unroll
    for (int r = 0; r < 8; ++r) {
#pragma unroll
      for (int off = 1; off < 16; off <<= 1)
        rsum[r] += __shfl_xor(rsum[r], off, 32);
      lrun[r] = lrun[r] * alpha[r] + rsum[r];
    }
#pragma unroll
    for (int n = 0; n < 4; ++n)
#pragma unroll
      for (int r = 0; r < 8; ++r) Oacc[n][r] *= alpha[r];

    // P (C layout) -> LDS -> A fragment; same-wave DS ops stay ordered
#pragma unroll
    for (int t = 0; t < 2; ++t)
#pragma unroll
      for (int r = 0; r < 8; ++r)
        P_lds[wave][(r + 8 * hi) * 32 + t * 16 + l16] = (half_t)p[t][r];

    v16h aP = load_a16(&P_lds[wave][0], 32, l16, 0, hi);
#pragma unroll
    for (int n = 0; n < 4; ++n) {
      v16h bV = load_b16(Vh, NP_, n * 16 + l16, ch * 32, hi);
      Oacc[n] = wmma16(aP, bV, Oacc[n]);
    }
  }

  // finalize: O /= l, store f16 [b][q][h*64 + d]
#pragma unroll
  for (int r = 0; r < 8; ++r) {
    float inv = 1.0f / lrun[r];
    int q = qt * 16 + r + 8 * hi;                        // < 592
#pragma unroll
    for (int n = 0; n < 4; ++n) {
      int d = n * 16 + l16;
      Of16[((size_t)(b * NP_ + q)) * C_ + h * HD_ + d] = (half_t)(Oacc[n][r] * inv);
    }
  }
}

// ----------------------------- output proj --------------------------------
// grid (QT_, B_), 256 threads (8 waves); wave owns 6 col-tiles of 16.
__global__ __launch_bounds__(256) void k_proj(const half_t* __restrict__ Of16,
                                              const half_t* __restrict__ PT,
                                              const float* __restrict__ proj_b,
                                              float* __restrict__ out) {
  const int qt = blockIdx.x, b = blockIdx.y;
  const int tid = threadIdx.x, wave = tid >> 5, lane = tid & 31;
  const int l16 = lane & 15, hi = lane >> 4;

  int row = b * NP_ + qt * 16 + l16;

  v8f acc[6];
#pragma unroll
  for (int j = 0; j < 6; ++j) acc[j] = (v8f){0,0,0,0,0,0,0,0};

  for (int kk = 0; kk < C_; kk += 32) {
    v16h a = load_a16(Of16, C_, row, kk, hi);
#pragma unroll
    for (int j = 0; j < 6; ++j) {
      int ct = wave * 6 + j;
      v16h bb = load_b16(PT, C_, ct * 16 + l16, kk, hi);
      acc[j] = wmma16(a, bb, acc[j]);
    }
  }
#pragma unroll
  for (int j = 0; j < 6; ++j) {
    int col = (wave * 6 + j) * 16 + l16;
    float bias = proj_b[col];
#pragma unroll
    for (int r = 0; r < 8; ++r) {
      int q = qt * 16 + r + 8 * hi;
      if (q < NTOK)
        out[((size_t)(b * NTOK + q)) * C_ + col] = acc[j][r] + bias;
    }
  }
}

// ------------------------------- launcher ---------------------------------

extern "C" void kernel_launch(void* const* d_in, const int* in_sizes, int n_in,
                              void* d_out, int out_size, void* d_ws, size_t ws_size,
                              hipStream_t stream) {
  (void)in_sizes; (void)n_in; (void)out_size; (void)ws_size;
  const float* x    = (const float*)d_in[0];
  const float* mks  = (const float*)d_in[1];
  const float* Wq   = (const float*)d_in[2];
  const float* Wk   = (const float*)d_in[3];
  const float* Wv   = (const float*)d_in[4];
  const float* mpj  = (const float*)d_in[5];
  const float* Pw   = (const float*)d_in[6];
  const float* pb   = (const float*)d_in[7];
  float* out = (float*)d_out;

  char* ws = (char*)d_ws;
  const size_t ROWS = (size_t)B_ * NTOK;                 // 4616
  half_t* xh   = (half_t*)ws;  ws += ROWS * C_ * sizeof(half_t);          // 7.09 MB
  half_t* WT   = (half_t*)ws;  ws += (size_t)QKVO_ * C_ * sizeof(half_t); // 1.77 MB
  half_t* PT   = (half_t*)ws;  ws += (size_t)C_ * C_ * sizeof(half_t);    // 1.18 MB
  half_t* Qws  = (half_t*)ws;  ws += (size_t)B_ * GH_ * NP_ * HD_ * sizeof(half_t);
  half_t* Kws  = (half_t*)ws;  ws += (size_t)B_ * GH_ * NP_ * HD_ * sizeof(half_t);
  half_t* Vt   = (half_t*)ws;  ws += (size_t)B_ * LH_ * HD_ * NP_ * sizeof(half_t);
  half_t* Of16 = (half_t*)ws;  ws += (size_t)B_ * NP_ * C_ * sizeof(half_t);
  // total ~28.7 MB

  int nx = (int)(ROWS * C_);
  k_cvt_x   <<<(nx + 255) / 256, 256, 0, stream>>>(x, xh, nx);
  k_build_wT<<<(QKVO_ * C_ + 255) / 256, 256, 0, stream>>>(Wq, Wk, Wv, WT);
  k_build_pT<<<(C_ * C_ + 255) / 256, 256, 0, stream>>>(Pw, PT);

  k_qkv <<<dim3(QTF_, B_), 256, 0, stream>>>(xh, WT, Qws, Kws, Vt);
  k_attn<<<dim3(QT_,  B_), 384, 0, stream>>>(Qws, Kws, Vt, mks, mpj, Of16);
  k_proj<<<dim3(QT_,  B_), 256, 0, stream>>>(Of16, PT, pb, out);
}
